// SparseBottleneck_77713138253908
// MI455X (gfx1250) — matrix operand order repaired and checked by reading, not compiled, not run on hardware
//
#include <hip/hip_runtime.h>
#include <hip/hip_bf16.h>
#include <math.h>

// Problem constants (from reference): B,C,H,T,K
#define Bb 32
#define Cc 128
#define Hh 512
#define Tt 8192
#define Kk 512

typedef __bf16 bf16;
typedef __attribute__((ext_vector_type(16))) __bf16 bf16x16;
typedef __attribute__((ext_vector_type(8)))  __bf16 bf16x8;
typedef __attribute__((ext_vector_type(8)))  float  f32x8;

// Monotonic float -> uint32 remap (order preserving), and inverse (exact bijection).
__device__ __forceinline__ unsigned mono_key(float f) {
    unsigned u = __float_as_uint(f);
    return u ^ (unsigned)(((int)u >> 31) | 0x80000000);
}
__device__ __forceinline__ float mono_unkey(unsigned k) {
    unsigned u = (k & 0x80000000u) ? (k ^ 0x80000000u) : ~k;
    return __uint_as_float(u);
}

// ---------------------------------------------------------------------------
// Kernel 1: out[b,c,t] = down_b[c]  (dense baseline of the output)
// ---------------------------------------------------------------------------
__global__ __launch_bounds__(256) void k_fill(const float* __restrict__ down_b,
                                              float* __restrict__ out) {
    size_t g = (size_t)blockIdx.x * 256 + threadIdx.x;   // float4 index
    int c = (int)((g * 4 / Tt) % Cc);
    float v = down_b[c];
    ((float4*)out)[g] = make_float4(v, v, v, v);
}

// ---------------------------------------------------------------------------
// Kernel 2: sal[b,h,t] = sal_w[h,:]·x[b,:,t] + sal_b[h]   (WMMA bf16 GEMM)
// Block tile 64(H) x 128(T), K = C = 128 in 4 steps of 32.
// 8 waves: 4 in M x 2 in N; each wave: 16x16 WMMA over 4 N-subtiles.
// Epilogue fuses softmax stats (two-level online softmax) + radix-1 histogram,
// eliminating a full 537MB re-read of sal.
// ---------------------------------------------------------------------------
#define LDA 136   // padded LDS row stride in bf16 elements (272B = 16B-aligned)

__global__ __launch_bounds__(256) void k_gemm_sal(
    const float* __restrict__ x, const float* __restrict__ sal_w,
    const float* __restrict__ sal_b, float* __restrict__ sal,
    unsigned* __restrict__ maxkey, unsigned* __restrict__ hist1,
    float* __restrict__ blockmax, float* __restrict__ blocksum)
{
    __shared__ bf16 As[64 * LDA];     // A tile: [h 0..63][c 0..127]
    __shared__ bf16 Bs[128 * LDA];    // B tile transposed: [t 0..127][c 0..127]
    __shared__ float wred[8];
    __shared__ float bmax_sh;

    const int tid = threadIdx.x;
    const int b  = blockIdx.z;
    const int h0 = blockIdx.y * 64;
    const int t0 = blockIdx.x * 128;

    // Stage A: sal_w[h0+row][col] (row-major [H,C]) -> bf16 LDS
    {
        const float4* src = (const float4*)(sal_w + (size_t)h0 * Cc);
        #pragma unroll
        for (int i = 0; i < 8; ++i) {
            int j = tid + i * 256;                 // 0..2047 float4
            int row = j >> 5, col4 = (j & 31) << 2;
            float4 v = src[j];
            bf16* dst = &As[row * LDA + col4];
            dst[0] = (bf16)v.x; dst[1] = (bf16)v.y;
            dst[2] = (bf16)v.z; dst[3] = (bf16)v.w;
        }
    }
    // Stage B (transposed): Bs[t][c] = x[b][c][t0+t]
    {
        #pragma unroll
        for (int i = 0; i < 16; ++i) {
            int j = tid + i * 256;                 // 0..4095 float4
            int c = j >> 5, tt4 = (j & 31) << 2;
            float4 v = *(const float4*)(x + ((size_t)b * Cc + c) * Tt + t0 + tt4);
            Bs[(tt4 + 0) * LDA + c] = (bf16)v.x;
            Bs[(tt4 + 1) * LDA + c] = (bf16)v.y;
            Bs[(tt4 + 2) * LDA + c] = (bf16)v.z;
            Bs[(tt4 + 3) * LDA + c] = (bf16)v.w;
        }
    }
    __syncthreads();

    const int w = tid >> 5, lane = tid & 31;
    const int wm = w & 3;        // M strip:  h0 + wm*16
    const int wn = w >> 2;       // N half:   t0 + wn*64
    const int m  = lane & 15;
    const int hiHalf = lane >> 4;

    f32x8 acc[4];
    #pragma unroll
    for (int s = 0; s < 4; ++s) acc[s] = (f32x8){0.f,0.f,0.f,0.f,0.f,0.f,0.f,0.f};

    #pragma unroll
    for (int kc = 0; kc < 128; kc += 32) {
        // A fragment (16x32 bf16): lanes<16 hold K {0..7,16..23}, lanes>=16 {8..15,24..31}
        bf16x16 af;
        const bf16* pa = &As[(wm * 16 + m) * LDA + kc + hiHalf * 8];
        ((bf16x8*)&af)[0] = *(const bf16x8*)pa;          // K base .. +7
        ((bf16x8*)&af)[1] = *(const bf16x8*)(pa + 16);   // K base+16 .. +23
        #pragma unroll
        for (int s = 0; s < 4; ++s) {
            // B fragment (32x16 bf16): lane n=lane&15, lanes<16 K {0..7,16..23}, lanes>=16 {8..15,24..31}
            const bf16* pb = &Bs[(wn * 64 + s * 16 + m) * LDA + kc + hiHalf * 8];
            bf16x16 bfr;
            ((bf16x8*)&bfr)[0] = *(const bf16x8*)pb;
            ((bf16x8*)&bfr)[1] = *(const bf16x8*)(pb + 16);
            acc[s] = __builtin_amdgcn_wmma_f32_16x16x32_bf16(
                false, af, false, bfr, (short)0, acc[s], false, false);
        }
    }

    // ---- Fused epilogue: bias + store + hist1 + two-level softmax stats ----
    // Bias values for this thread's 8 output rows.
    float bias[8];
    #pragma unroll
    for (int r = 0; r < 8; ++r) {
        int hm = hiHalf ? (r + 8) : r;
        bias[r] = sal_b[h0 + wm * 16 + hm];
    }

    unsigned* hb = hist1 + (size_t)b * 65536;
    float lmax = -3.4e38f;
    #pragma unroll
    for (int s = 0; s < 4; ++s) {
        int t = t0 + wn * 64 + s * 16 + m;
        #pragma unroll
        for (int r = 0; r < 8; ++r) {
            int hm = hiHalf ? (r + 8) : r;          // C/D layout: lanes>=16 hold M=8..15
            int h = h0 + wm * 16 + hm;
            float v = acc[s][r] + bias[r];
            sal[((size_t)b * Hh + h) * Tt + t] = v;
            atomicAdd(&hb[mono_key(v) >> 16], 1u);
            lmax = fmaxf(lmax, v);
        }
    }
    // Block max: wave shfl-reduce, then cross-wave via LDS.
    #pragma unroll
    for (int off = 16; off > 0; off >>= 1)
        lmax = fmaxf(lmax, __shfl_xor(lmax, off, 32));
    if (lane == 0) wred[w] = lmax;
    __syncthreads();
    if (tid == 0) {
        float mxx = wred[0];
        #pragma unroll
        for (int i = 1; i < 8; ++i) mxx = fmaxf(mxx, wred[i]);
        bmax_sh = mxx;
    }
    __syncthreads();
    const float bmax = bmax_sh;

    // Block-shifted sum of exponentials (values still in acc registers).
    float ssum = 0.f;
    #pragma unroll
    for (int s = 0; s < 4; ++s)
        #pragma unroll
        for (int r = 0; r < 8; ++r)
            ssum += __expf(acc[s][r] + bias[r] - bmax);
    #pragma unroll
    for (int off = 16; off > 0; off >>= 1) ssum += __shfl_xor(ssum, off, 32);
    __syncthreads();               // wred reuse hazard
    if (lane == 0) wred[w] = ssum;
    __syncthreads();
    if (tid == 0) {
        float tot = 0.f;
        #pragma unroll
        for (int i = 0; i < 8; ++i) tot += wred[i];
        int slot = ((b * 8) + blockIdx.y) * 64 + blockIdx.x;   // 512 blocks/batch
        blockmax[slot] = bmax;
        blocksum[slot] = tot;
        atomicMax(&maxkey[b], mono_key(bmax));
    }
}

// ---------------------------------------------------------------------------
// Kernel 3: combine per-block (max, sum) -> per-batch softmax denominator.
//   sumexp[b] = sum_i blocksum[i] * exp(blockmax[i] - globalmax)
// ---------------------------------------------------------------------------
__global__ __launch_bounds__(512) void k_finalize(
    const float* __restrict__ blockmax, const float* __restrict__ blocksum,
    const unsigned* __restrict__ maxkey, float* __restrict__ sumexp)
{
    const int b = blockIdx.x;
    const float gmax = mono_unkey(maxkey[b]);
    const int i = b * 512 + threadIdx.x;
    float p = blocksum[i] * __expf(blockmax[i] - gmax);
    __shared__ float red[512];
    red[threadIdx.x] = p; __syncthreads();
    for (int st = 256; st > 0; st >>= 1) {
        if (threadIdx.x < st) red[threadIdx.x] += red[threadIdx.x + st];
        __syncthreads();
    }
    if (threadIdx.x == 0) sumexp[b] = red[0];
}

// ---------------------------------------------------------------------------
// Kernel 4: radix pass 1 — find top-16-bit bin containing the K-th largest
// ---------------------------------------------------------------------------
__global__ __launch_bounds__(1024) void k_pick1(
    const unsigned* __restrict__ hist1,
    unsigned* __restrict__ bin1, unsigned* __restrict__ cnt1)
{
    const int b = blockIdx.x;
    const unsigned* hb = hist1 + (size_t)b * 65536;
    __shared__ unsigned csum[1024];
    __shared__ unsigned selChunk, selAbove;
    const int c0 = threadIdx.x * 64;
    unsigned s = 0;
    for (int i = 0; i < 64; ++i) s += hb[c0 + i];
    csum[threadIdx.x] = s;
    __syncthreads();
    if (threadIdx.x == 0) {
        unsigned run = 0; unsigned sel = 0, above = 0;
        for (int i = 1023; i >= 0; --i) {
            if (run + csum[i] >= (unsigned)Kk) { sel = i; above = run; break; }
            run += csum[i];
        }
        selChunk = sel; selAbove = above;
    }
    __syncthreads();
    if (threadIdx.x == (int)selChunk) {
        unsigned run = selAbove;
        for (int i = 63; i >= 0; --i) {
            unsigned h = hb[c0 + i];
            if (run + h >= (unsigned)Kk) { bin1[b] = (unsigned)(c0 + i); cnt1[b] = run; break; }
            run += h;
        }
    }
}

// ---------------------------------------------------------------------------
// Kernel 5: radix pass 2 histogram — low 16 bits of keys within bin1
// ---------------------------------------------------------------------------
__global__ __launch_bounds__(256) void k_hist2(
    const float* __restrict__ sal, const unsigned* __restrict__ bin1,
    unsigned* __restrict__ hist2)
{
    const int b = blockIdx.y;
    const unsigned b1 = bin1[b];
    unsigned* hb = hist2 + (size_t)b * 65536;
    const size_t base = (size_t)b * Hh * Tt + (size_t)blockIdx.x * 4096;
    #pragma unroll
    for (int k2 = 0; k2 < 4; ++k2) {
        float4 v = *(const float4*)(sal + base + k2 * 1024 + threadIdx.x * 4);
        float a[4] = {v.x, v.y, v.z, v.w};
        #pragma unroll
        for (int j = 0; j < 4; ++j) {
            unsigned key = mono_key(a[j]);
            if ((key >> 16) == b1) atomicAdd(&hb[key & 0xFFFFu], 1u);
        }
    }
}

// ---------------------------------------------------------------------------
// Kernel 6: radix pass 2 select — exact 32-bit threshold key + tie budget
// ---------------------------------------------------------------------------
__global__ __launch_bounds__(1024) void k_pick2(
    const unsigned* __restrict__ hist2, const unsigned* __restrict__ bin1,
    const unsigned* __restrict__ cnt1,
    unsigned* __restrict__ thrkey, unsigned* __restrict__ need)
{
    const int b = blockIdx.x;
    const unsigned* hb = hist2 + (size_t)b * 65536;
    const unsigned target = (unsigned)Kk - cnt1[b];
    __shared__ unsigned csum[1024];
    __shared__ unsigned selChunk, selAbove;
    const int c0 = threadIdx.x * 64;
    unsigned s = 0;
    for (int i = 0; i < 64; ++i) s += hb[c0 + i];
    csum[threadIdx.x] = s;
    __syncthreads();
    if (threadIdx.x == 0) {
        unsigned run = 0; unsigned sel = 0, above = 0;
        for (int i = 1023; i >= 0; --i) {
            if (run + csum[i] >= target) { sel = i; above = run; break; }
            run += csum[i];
        }
        selChunk = sel; selAbove = above;
    }
    __syncthreads();
    if (threadIdx.x == (int)selChunk) {
        unsigned run = selAbove;
        for (int i = 63; i >= 0; --i) {
            unsigned h = hb[c0 + i];
            if (run + h >= target) {
                thrkey[b] = (bin1[b] << 16) | (unsigned)(c0 + i);
                need[b]   = target - run;
                break;
            }
            run += h;
        }
    }
}

// ---------------------------------------------------------------------------
// Kernel 7: compact the top-K (index, logit) pairs per batch
// ---------------------------------------------------------------------------
__global__ __launch_bounds__(256) void k_compact(
    const float* __restrict__ sal, const unsigned* __restrict__ thrkey,
    const unsigned* __restrict__ need, unsigned* __restrict__ ties,
    unsigned* __restrict__ csel, unsigned* __restrict__ idxl,
    float* __restrict__ vall)
{
    const int b = blockIdx.y;
    const unsigned thr = thrkey[b];
    const unsigned nd  = need[b];
    const size_t bbase = (size_t)b * Hh * Tt;
    const size_t blk   = (size_t)blockIdx.x * 4096;
    #pragma unroll
    for (int k2 = 0; k2 < 4; ++k2) {
        size_t li0 = blk + (size_t)k2 * 1024 + (size_t)threadIdx.x * 4;
        float4 v = *(const float4*)(sal + bbase + li0);
        float a[4] = {v.x, v.y, v.z, v.w};
        #pragma unroll
        for (int j = 0; j < 4; ++j) {
            unsigned key = mono_key(a[j]);
            bool take = key > thr;
            if (key == thr) take = (atomicAdd(&ties[b], 1u) < nd);
            if (take) {
                unsigned p = atomicAdd(&csel[b], 1u);
                if (p < (unsigned)Kk) {
                    idxl[(size_t)b * Kk + p] = (unsigned)(li0 + j);
                    vall[(size_t)b * Kk + p] = a[j];
                }
            }
        }
    }
}

// ---------------------------------------------------------------------------
// Kernel 8: sparse scatter — one wave per selected (h,t):
//   p   = exp(sal - max)/sumexp
//   sig = up_w[h,:]·x[b,:,t] + up_b[h]   (recomputed, 128 MACs/wave)
//   out[b,:,t] += down_w[:,h] * (sig * p)
// ---------------------------------------------------------------------------
__global__ __launch_bounds__(256) void k_scatter(
    const float* __restrict__ x, const float* __restrict__ up_w,
    const float* __restrict__ up_b, const float* __restrict__ down_w,
    const unsigned* __restrict__ maxkey, const float* __restrict__ sumexp,
    const unsigned* __restrict__ csel, const unsigned* __restrict__ idxl,
    const float* __restrict__ vall, float* __restrict__ out)
{
    const int gw   = blockIdx.x * 8 + (threadIdx.x >> 5);
    const int lane = threadIdx.x & 31;
    const int b = gw / Kk, e = gw % Kk;
    if ((unsigned)e >= csel[b]) return;

    const unsigned li = idxl[(size_t)b * Kk + e];
    const int h = (int)(li / Tt), t = (int)(li % Tt);
    const float mx = mono_unkey(maxkey[b]);
    const float p  = __expf(vall[(size_t)b * Kk + e] - mx) / sumexp[b];

    const int c4 = lane * 4;
    float sig = 0.f;
    #pragma unroll
    for (int j = 0; j < 4; ++j)
        sig += up_w[h * Cc + c4 + j] * x[((size_t)b * Cc + c4 + j) * Tt + t];
    #pragma unroll
    for (int off = 16; off > 0; off >>= 1) sig += __shfl_xor(sig, off, 32);

    const float g = (sig + up_b[h]) * p;
    #pragma unroll
    for (int j = 0; j < 4; ++j) {
        int c = c4 + j;
        atomicAdd(&out[((size_t)b * Cc + c) * Tt + t], down_w[c * Hh + h] * g);
    }
}

// ---------------------------------------------------------------------------
extern "C" void kernel_launch(void* const* d_in, const int* in_sizes, int n_in,
                              void* d_out, int out_size, void* d_ws, size_t ws_size,
                              hipStream_t stream)
{
    (void)in_sizes; (void)n_in; (void)out_size; (void)ws_size;
    const float* x      = (const float*)d_in[0];
    const float* up_w   = (const float*)d_in[1];
    const float* up_b   = (const float*)d_in[2];
    const float* sal_w  = (const float*)d_in[3];
    const float* sal_b  = (const float*)d_in[4];
    const float* down_w = (const float*)d_in[5];
    const float* down_b = (const float*)d_in[6];
    float* out = (float*)d_out;

    char* ws = (char*)d_ws;
    size_t off = 0;
    auto take = [&](size_t bytes) { void* p = ws + off; off += bytes; return p; };
    float*    sal      = (float*)   take((size_t)Bb * Hh * Tt * sizeof(float));
    // ---- zero-initialized region (contiguous) ----
    unsigned* hist1    = (unsigned*)take((size_t)Bb * 65536 * sizeof(unsigned));
    unsigned* hist2    = (unsigned*)take((size_t)Bb * 65536 * sizeof(unsigned));
    unsigned* maxkey   = (unsigned*)take(Bb * sizeof(unsigned));
    unsigned* ties     = (unsigned*)take(Bb * sizeof(unsigned));
    unsigned* csel     = (unsigned*)take(Bb * sizeof(unsigned));
    // ---- written-before-read region ----
    float*    sumexp   = (float*)   take(Bb * sizeof(float));
    unsigned* bin1     = (unsigned*)take(Bb * sizeof(unsigned));
    unsigned* cnt1     = (unsigned*)take(Bb * sizeof(unsigned));
    unsigned* thrkey   = (unsigned*)take(Bb * sizeof(unsigned));
    unsigned* need     = (unsigned*)take(Bb * sizeof(unsigned));
    float*    blockmax = (float*)   take((size_t)Bb * 512 * sizeof(float));
    float*    blocksum = (float*)   take((size_t)Bb * 512 * sizeof(float));
    unsigned* idxl     = (unsigned*)take((size_t)Bb * Kk * sizeof(unsigned));
    float*    vall     = (float*)   take((size_t)Bb * Kk * sizeof(float));

    size_t zbytes = (size_t)Bb * 65536 * sizeof(unsigned) * 2 + (size_t)Bb * 3 * sizeof(unsigned);
    hipMemsetAsync(hist1, 0, zbytes, stream);

    k_fill    <<<(Bb * Cc * Tt / 4) / 256, 256, 0, stream>>>(down_b, out);
    k_gemm_sal<<<dim3(Tt / 128, Hh / 64, Bb), 256, 0, stream>>>(
        x, sal_w, sal_b, sal, maxkey, hist1, blockmax, blocksum);
    k_finalize<<<Bb, 512, 0, stream>>>(blockmax, blocksum, maxkey, sumexp);
    k_pick1   <<<Bb, 1024, 0, stream>>>(hist1, bin1, cnt1);
    k_hist2   <<<dim3(1024, Bb), 256, 0, stream>>>(sal, bin1, hist2);
    k_pick2   <<<Bb, 1024, 0, stream>>>(hist2, bin1, cnt1, thrkey, need);
    k_compact <<<dim3(1024, Bb), 256, 0, stream>>>(sal, thrkey, need, ties, csel, idxl, vall);
    k_scatter <<<(Bb * Kk) / 8, 256, 0, stream>>>(x, up_w, up_b, down_w,
                                                  maxkey, sumexp, csel, idxl, vall, out);
}
